// StackLayerNet_86165633892442
// MI455X (gfx1250) — matrix operand order repaired and checked by reading, not compiled, not run on hardware
//
#include <hip/hip_runtime.h>
#include <math.h>

#define B_ 8          // batch
#define NPT 2048      // points per cloud

typedef __attribute__((ext_vector_type(8)))  _Float16 v8h;
typedef __attribute__((ext_vector_type(16))) _Float16 v16h;
typedef __attribute__((ext_vector_type(8)))  float    v8f;

// ===================== WMMA GEMM (CDNA5, wave32) =====================
// C[M,N] (fp32) = A[M,Kp] (fp16, lda=Kp) @ Bt (fp16, N-major: Bt[n][k], ldb=Kp).
// Kp padded to mult of 32 and zero-filled; M multiple of 16 (grid exact);
// N guarded only at store. One wave per block computes a 16x32 tile
// (two 16x16 accumulators sharing the A fragment).
// Fragment layouts used here (wave32, same contiguous k order for A and B):
//   A: lane m=lane&15 -> row; element i -> k = i + 16*(lane>>4)
//      => two contiguous 8-half runs  -> 2x b128 loads
//   B: lane n=lane&15 -> col; element i -> k = i + 16*(lane>>4)
//      => one contiguous 16-half run  -> 1x v16h load
//   C/D: col = lane&15 (+tn), row = r + 8*(lane>>4) (+tm)
// EPI: 0=none, 1=+bias, 2=relu(g*(x+b)+be), 3=sigmoid(x+b)
template<int EPI>
__global__ __launch_bounds__(32)
void wmma_gemm_kernel(const _Float16* __restrict__ A, const _Float16* __restrict__ Bt,
                      const float* __restrict__ bias, const float* __restrict__ gam,
                      const float* __restrict__ bet, float* __restrict__ C,
                      int N, int Kp, int lda, int ldb, int ldc,
                      long long sA, long long sB, long long sC)
{
  const int lane = threadIdx.x;
  const int tm = blockIdx.y << 4;
  const int tn = blockIdx.x << 5;
  const int m  = lane & 15;
  const int kh = lane >> 4;
  const _Float16* Ar  = A  + (long long)blockIdx.z * sA + (long long)(tm + m) * lda;
  const _Float16* Br0 = Bt + (long long)blockIdx.z * sB + (long long)(tn + m) * ldb;
  const _Float16* Br1 = Br0 + 16LL * ldb;
  C += (long long)blockIdx.z * sC;

  v8f acc0 = {0.f,0.f,0.f,0.f,0.f,0.f,0.f,0.f};
  v8f acc1 = {0.f,0.f,0.f,0.f,0.f,0.f,0.f,0.f};

  const int a_off = kh << 4;   // 16*half
  const int b_off = kh << 4;   // 16*half
  for (int k0 = 0; k0 < Kp; k0 += 32) {
    __builtin_prefetch(Ar + k0 + 64, 0, 1);          // speculative, fault-free
    v8h  al  = *(const v8h*)(Ar + k0 + a_off);
    v8h  ah  = *(const v8h*)(Ar + k0 + a_off + 8);
    v16h bf0 = *(const v16h*)(Br0 + k0 + b_off);
    v16h bf1 = *(const v16h*)(Br1 + k0 + b_off);
    v16h af  = __builtin_shufflevector(al, ah, 0,1,2,3,4,5,6,7,8,9,10,11,12,13,14,15);
    acc0 = __builtin_amdgcn_wmma_f32_16x16x32_f16(false, af, false, bf0, (short)0, acc0, false, false);
    acc1 = __builtin_amdgcn_wmma_f32_16x16x32_f16(false, af, false, bf1, (short)0, acc1, false, false);
  }

  const int col0 = tn + m;
  const int col1 = col0 + 16;
#pragma unroll
  for (int r = 0; r < 8; ++r) {
    const int row = tm + r + (kh << 3);
    float v0 = acc0[r], v1 = acc1[r];
    if (col0 < N) {
      float v = v0;
      if (EPI == 1) v += bias[col0];
      if (EPI == 2) v = fmaxf(gam[col0] * (v + bias[col0]) + bet[col0], 0.f);
      if (EPI == 3) { v += bias[col0]; v = 1.f / (1.f + __expf(-v)); }
      C[(long long)row * ldc + col0] = v;
    }
    if (col1 < N) {
      float v = v1;
      if (EPI == 1) v += bias[col1];
      if (EPI == 2) v = fmaxf(gam[col1] * (v + bias[col1]) + bet[col1], 0.f);
      if (EPI == 3) { v += bias[col1]; v = 1.f / (1.f + __expf(-v)); }
      C[(long long)row * ldc + col1] = v;
    }
  }
}

// ===================== packing kernels =====================

// fp32 (rows x cols, srcld) -> fp16 (rows x dstld), zero-padding cols [cols, dstld)
__global__ void pack_a_kernel(const float* __restrict__ src, int srcld, int cols,
                              _Float16* __restrict__ dst, int dstld, long long total)
{
  long long t = (long long)blockIdx.x * 256 + threadIdx.x;
  if (t >= total) return;
  long long r = t / dstld; int c = (int)(t % dstld);
  dst[t] = (c < cols) ? (_Float16)src[r * srcld + c] : (_Float16)0.f;
}

// W[K][N] fp32 -> Bt[Np][Kp] fp16 (N-major, zero-padded)
__global__ void pack_wt_kernel(const float* __restrict__ W, int K, int N,
                               _Float16* __restrict__ dst, int Kp, long long total)
{
  long long t = (long long)blockIdx.x * 256 + threadIdx.x;
  if (t >= total) return;
  long long n = t / Kp; int k = (int)(t % Kp);
  dst[t] = (k < K && n < (long long)N) ? (_Float16)W[(long long)k * N + n] : (_Float16)0.f;
}

// V[b][2048][128] fp32 -> Vt[b][128][2048] fp16
__global__ void pack_vt_kernel(const float* __restrict__ V, _Float16* __restrict__ dst, long long total)
{
  long long t = (long long)blockIdx.x * 256 + threadIdx.x;
  if (t >= total) return;
  int b = (int)(t / (128LL * NPT));
  long long rem = t % (128LL * NPT);
  int c = (int)(rem / NPT);
  int r = (int)(rem % NPT);
  dst[t] = (_Float16)V[((long long)b * NPT + r) * 128 + c];
}

// ===================== utility kernels (fp32) =====================

__global__ void fps_kernel(const float* __restrict__ xyz, int ldx, int N, int npoint, int* __restrict__ out)
{
  __shared__ float dist[2048];
  __shared__ float rv[256];
  __shared__ int   ri[256];
  __shared__ int   sfar;
  const int b = blockIdx.x, tid = threadIdx.x;
  const float* X = xyz + (long long)b * N * ldx;
  for (int n = tid; n < N; n += 256) dist[n] = 1e10f;
  if (tid == 0) sfar = 0;
  __syncthreads();
  for (int it = 0; it < npoint; ++it) {
    const int far = sfar;
    if (tid == 0) out[b * npoint + it] = far;
    const float cx = X[far*ldx], cy = X[far*ldx+1], cz = X[far*ldx+2];
    float bm = -1.f; int bi = 0;
    for (int n = tid; n < N; n += 256) {
      float dx = X[n*ldx]-cx, dy = X[n*ldx+1]-cy, dz = X[n*ldx+2]-cz;
      float d  = dx*dx + dy*dy + dz*dz;
      float dm = fminf(dist[n], d);
      dist[n] = dm;
      if (dm > bm) { bm = dm; bi = n; }
    }
    rv[tid] = bm; ri[tid] = bi;
    __syncthreads();
    for (int s = 128; s > 0; s >>= 1) {
      if (tid < s) {
        if (rv[tid+s] > rv[tid] || (rv[tid+s] == rv[tid] && ri[tid+s] < ri[tid])) {
          rv[tid] = rv[tid+s]; ri[tid] = ri[tid+s];
        }
      }
      __syncthreads();
    }
    if (tid == 0) sfar = ri[0];
    __syncthreads();
  }
}

__global__ void gather_xyz_kernel(const float* __restrict__ xyz, int ldx, int N, int S,
                                  const int* __restrict__ idx, float* __restrict__ out)
{
  int t = blockIdx.x * 256 + threadIdx.x;
  if (t >= B_ * S) return;
  int b = t / S;
  int id = idx[t];
  const float* X = xyz + ((long long)b * N + id) * ldx;
  out[t*3+0] = X[0]; out[t*3+1] = X[1]; out[t*3+2] = X[2];
}

__global__ void ballq_kernel(const float* __restrict__ xyz, int ldx, const float* __restrict__ nxyz,
                             int N, int S, int ns, float r2, int* __restrict__ gidx)
{
  int t = blockIdx.x * 256 + threadIdx.x;
  if (t >= B_ * S) return;
  int b = t / S;
  const float* X = xyz + (long long)b * N * ldx;
  float cx = nxyz[t*3], cy = nxyz[t*3+1], cz = nxyz[t*3+2];
  int* g = gidx + (long long)t * ns;
  int cnt = 0, first = -1;
  for (int n = 0; n < N; ++n) {
    float dx = X[n*ldx]-cx, dy = X[n*ldx+1]-cy, dz = X[n*ldx+2]-cz;
    float d = dx*dx + dy*dy + dz*dz;
    if (d <= r2) {
      if (first < 0) first = n;
      if (cnt < ns) g[cnt++] = n;
    }
  }
  if (first < 0) first = 0;
  for (; cnt < ns; ++cnt) g[cnt] = first;
}

__global__ void group_kernel(const float* __restrict__ xyz, int ldx,
                             const float* __restrict__ pts, int ldp, int Cp,
                             const float* __restrict__ nxyz, const int* __restrict__ gidx,
                             int N, int S, int ns, float* __restrict__ out)
{
  long long t = (long long)blockIdx.x * 256 + threadIdx.x;
  long long total = (long long)B_ * S * ns;
  if (t >= total) return;
  int b = (int)(t / ((long long)S * ns));
  int rem = (int)(t % ((long long)S * ns));
  int s = rem / ns;
  int id = gidx[t];
  const float* X = xyz + (long long)b * N * ldx;
  const float* P = pts + (long long)b * N * ldp;
  const float* c = nxyz + ((long long)b * S + s) * 3;
  float* o = out + t * (3 + Cp);
  o[0] = X[id*ldx+0] - c[0];
  o[1] = X[id*ldx+1] - c[1];
  o[2] = X[id*ldx+2] - c[2];
  for (int k = 0; k < Cp; ++k) o[3+k] = P[(long long)id*ldp + k];
}

__global__ void maxpool_kernel(const float* __restrict__ h, long long rows, int ns, int C, float* __restrict__ out)
{
  long long t = (long long)blockIdx.x * 256 + threadIdx.x;
  if (t >= rows * C) return;
  long long s = t / C; int c = (int)(t % C);
  float m = -1e30f;
  for (int j = 0; j < ns; ++j) m = fmaxf(m, h[((long long)s * ns + j) * C + c]);
  out[t] = m;
}

__global__ void se_scale_kernel(const float* __restrict__ x, int n, int C, int h,
                                const float* __restrict__ w1, const float* __restrict__ b1,
                                const float* __restrict__ w2, const float* __restrict__ b2,
                                float* __restrict__ scale)
{
  __shared__ float sm[1024];
  __shared__ float sy[64];
  int b = blockIdx.x, tid = threadIdx.x;
  const float* X = x + (long long)b * n * C;
  for (int c = tid; c < C; c += 256) {
    float s = 0.f;
    for (int i = 0; i < n; ++i) s += X[(long long)i * C + c];
    sm[c] = s / (float)n;
  }
  __syncthreads();
  if (tid < h) {
    float y = b1[tid];
    for (int c = 0; c < C; ++c) y += sm[c] * w1[c * h + tid];
    sy[tid] = fmaxf(y, 0.f);
  }
  __syncthreads();
  for (int c = tid; c < C; c += 256) {
    float z = b2[c];
    for (int j = 0; j < h; ++j) z += sy[j] * w2[j * C + c];
    scale[b * C + c] = 1.f / (1.f + __expf(-z));
  }
}

__global__ void se_apply_kernel(float* __restrict__ x, const float* __restrict__ scale,
                                int n, int C, long long total)
{
  long long t = (long long)blockIdx.x * 256 + threadIdx.x;
  if (t >= total) return;
  int c = (int)(t % C);
  int b = (int)(t / ((long long)n * C));
  x[t] *= scale[b * C + c];
}

__global__ void knn3_kernel(const float* __restrict__ xyz1, int ld1, int N1,
                            const float* __restrict__ xyz2, int N2,
                            const float* __restrict__ p2, int C2,
                            float* __restrict__ out, int ldo, int coff)
{
  int t = blockIdx.x * 256 + threadIdx.x;
  if (t >= B_ * N1) return;
  int b = t / N1, i = t % N1;
  const float* A = xyz1 + ((long long)b * N1 + i) * ld1;
  float px = A[0], py = A[1], pz = A[2];
  const float* X2 = xyz2 + (long long)b * N2 * 3;
  float d0 = 1e30f, d1 = 1e30f, d2 = 1e30f;
  int j0 = 0, j1 = 0, j2 = 0;
  for (int j = 0; j < N2; ++j) {
    float dx = X2[j*3]-px, dy = X2[j*3+1]-py, dz = X2[j*3+2]-pz;
    float d = dx*dx + dy*dy + dz*dz;
    if (d < d0)      { d2=d1; j2=j1; d1=d0; j1=j0; d0=d; j0=j; }
    else if (d < d1) { d2=d1; j2=j1; d1=d;  j1=j; }
    else if (d < d2) { d2=d;  j2=j; }
  }
  d0 = fmaxf(d0, 1e-10f); d1 = fmaxf(d1, 1e-10f); d2 = fmaxf(d2, 1e-10f);
  float w0 = 1.f/d0, w1 = 1.f/d1, w2 = 1.f/d2;
  float inv = 1.f / (w0 + w1 + w2);
  w0 *= inv; w1 *= inv; w2 *= inv;
  const float* P = p2 + (long long)b * N2 * C2;
  float* O = out + (long long)t * ldo + coff;
  for (int c = 0; c < C2; ++c)
    O[c] = w0 * P[(long long)j0*C2+c] + w1 * P[(long long)j1*C2+c] + w2 * P[(long long)j2*C2+c];
}

__global__ void copy_mat_kernel(const float* __restrict__ src, int lds_, int soff,
                                float* __restrict__ dst, int ldd, int doff,
                                long long rows, int cols)
{
  long long t = (long long)blockIdx.x * 256 + threadIdx.x;
  if (t >= rows * cols) return;
  long long r = t / cols; int c = (int)(t % cols);
  dst[r * ldd + doff + c] = src[r * lds_ + soff + c];
}

__global__ void broadcast_kernel(const float* __restrict__ src, float* __restrict__ dst,
                                 int ldd, int doff, int n, int C)
{
  long long t = (long long)blockIdx.x * 256 + threadIdx.x;
  long long total = (long long)B_ * n * C;
  if (t >= total) return;
  int b = (int)(t / ((long long)n * C));
  long long rem = t % ((long long)n * C);
  long long i = rem / C; int c = (int)(rem % C);
  dst[((long long)b * n + i) * ldd + doff + c] = src[b * C + c];
}

__global__ void lininterp_kernel(const float* __restrict__ src, int nin, int C,
                                 float* __restrict__ dst, int ldd, int doff, int outn)
{
  long long t = (long long)blockIdx.x * 256 + threadIdx.x;
  long long total = (long long)B_ * outn * C;
  if (t >= total) return;
  int b = (int)(t / ((long long)outn * C));
  long long rem = t % ((long long)outn * C);
  int i = (int)(rem / C); int c = (int)(rem % C);
  float pos = (float)i * ((float)(nin - 1) / (float)(outn - 1));
  int i0 = (int)floorf(pos);
  int i1 = (i0 + 1 < nin) ? i0 + 1 : nin - 1;
  float w = pos - (float)i0;
  const float* S = src + (long long)b * nin * C;
  dst[((long long)b * outn + i) * ldd + doff + c] =
      S[(long long)i0 * C + c] * (1.f - w) + S[(long long)i1 * C + c] * w;
}

__global__ void posenc_kernel(const float* __restrict__ feat, const float* __restrict__ x6,
                              const float* __restrict__ Wp, const float* __restrict__ bp,
                              float* __restrict__ f, long long total)
{
  long long t = (long long)blockIdx.x * 256 + threadIdx.x;
  if (t >= total) return;
  int c = (int)(t & 127);
  long long row = t >> 7;
  const float* xr = x6 + row * 6;
  f[t] = feat[t] + bp[c] + xr[0]*Wp[c] + xr[1]*Wp[128+c] + xr[2]*Wp[256+c];
}

__global__ void softmax_rows_kernel(float* __restrict__ s, int n)
{
  __shared__ float red[256];
  float* row = s + (long long)blockIdx.x * n;
  int tid = threadIdx.x;
  float m = -1e30f;
  for (int j = tid; j < n; j += 256) m = fmaxf(m, row[j]);
  red[tid] = m; __syncthreads();
  for (int st = 128; st > 0; st >>= 1) { if (tid < st) red[tid] = fmaxf(red[tid], red[tid+st]); __syncthreads(); }
  m = red[0]; __syncthreads();
  float sum = 0.f;
  for (int j = tid; j < n; j += 256) { float e = __expf(row[j] - m); row[j] = e; sum += e; }
  red[tid] = sum; __syncthreads();
  for (int st = 128; st > 0; st >>= 1) { if (tid < st) red[tid] += red[tid+st]; __syncthreads(); }
  float inv = 1.f / red[0];
  for (int j = tid; j < n; j += 256) row[j] *= inv;
}

__global__ void gamma_res_kernel(const float* __restrict__ a, const float* __restrict__ feat,
                                 const float* __restrict__ gamma, float* __restrict__ o, long long total)
{
  long long t = (long long)blockIdx.x * 256 + threadIdx.x;
  if (t >= total) return;
  o[t] = gamma[0] * a[t] + feat[t];
}

// ===================== host side =====================

static inline int blks(long long n) { return (int)((n + 255) / 256); }
static inline int pad32(int k) { return (k + 31) & ~31; }

extern "C" void kernel_launch(void* const* d_in, const int* in_sizes, int n_in,
                              void* d_out, int out_size, void* d_ws, size_t ws_size,
                              hipStream_t stream)
{
  (void)in_sizes; (void)n_in; (void)out_size; (void)ws_size;
  const float* x = (const float*)d_in[0];      // (8,2048,6)
  int pi = 1;
  auto nxt = [&]() { return (const float*)d_in[pi++]; };

  // ---- params in setup_inputs() insertion order ----
  const float *s1W[3], *s1b[3], *s1g[3], *s1e[3];
  for (int l = 0; l < 3; ++l) { s1W[l]=nxt(); s1b[l]=nxt(); s1g[l]=nxt(); s1e[l]=nxt(); }
  const float *s1w1=nxt(), *s1b1=nxt(), *s1w2=nxt(), *s1b2=nxt();
  const float *s2W[3], *s2b[3], *s2g[3], *s2e[3];
  for (int l = 0; l < 3; ++l) { s2W[l]=nxt(); s2b[l]=nxt(); s2g[l]=nxt(); s2e[l]=nxt(); }
  const float *s2w1=nxt(), *s2b1=nxt(), *s2w2=nxt(), *s2b2=nxt();
  const float *s3W[3], *s3b[3], *s3g[3], *s3e[3];
  for (int l = 0; l < 3; ++l) { s3W[l]=nxt(); s3b[l]=nxt(); s3g[l]=nxt(); s3e[l]=nxt(); }
  const float *s3w1=nxt(), *s3b1=nxt(), *s3w2=nxt(), *s3b2=nxt();
  const float *sgw1=nxt(), *sgb1=nxt(), *sgw2=nxt(), *sgb2=nxt();   // se_global
  const float *smw1=nxt(), *smb1=nxt(), *smw2=nxt(), *smb2=nxt();   // se_mid
  const float *f3W[2], *f3b[2], *f3g[2], *f3e[2];
  for (int l = 0; l < 2; ++l) { f3W[l]=nxt(); f3b[l]=nxt(); f3g[l]=nxt(); f3e[l]=nxt(); }
  const float *f3w1=nxt(), *f3b1=nxt(), *f3w2=nxt(), *f3b2=nxt();
  const float *f2W[3], *f2b[3], *f2g[3], *f2e[3];
  for (int l = 0; l < 3; ++l) { f2W[l]=nxt(); f2b[l]=nxt(); f2g[l]=nxt(); f2e[l]=nxt(); }
  const float *f2w1=nxt(), *f2b1=nxt(), *f2w2=nxt(), *f2b2=nxt();
  const float *f1W[3], *f1b[3], *f1g[3], *f1e[3];
  for (int l = 0; l < 3; ++l) { f1W[l]=nxt(); f1b[l]=nxt(); f1g[l]=nxt(); f1e[l]=nxt(); }
  const float *f1w1=nxt(), *f1b1=nxt(), *f1w2=nxt(), *f1b2=nxt();
  const float *Wp=nxt(), *bp=nxt(), *Wq=nxt(), *Wk=nxt(), *Wv=nxt(), *gamma=nxt();
  const float *fuW=nxt(), *fub=nxt(), *fug=nxt(), *fue=nxt();
  const float *shW=nxt(), *shb=nxt(), *shg=nxt(), *she=nxt();
  const float *hW1[4], *hb1[4], *hg1[4], *he1[4], *hW2[4], *hb2[4];
  for (int hd = 0; hd < 4; ++hd) { hW1[hd]=nxt(); hb1[hd]=nxt(); hg1[hd]=nxt(); he1[hd]=nxt(); hW2[hd]=nxt(); hb2[hd]=nxt(); }

  // ---- workspace carve ----
  size_t off = 0; char* wsb = (char*)d_ws;
  auto allocf = [&](long long n) { float*    p = (float*)   (wsb + off); off += (size_t)(((n*4 + 255)/256)*256); return p; };
  auto alloch = [&](long long n) { _Float16* p = (_Float16*)(wsb + off); off += (size_t)(((n*2 + 255)/256)*256); return p; };
  auto alloci = [&](long long n) { int*      p = (int*)     (wsb + off); off += (size_t)(((n*4 + 255)/256)*256); return p; };
  float*    ping    = allocf(131072LL * 128);
  float*    pong    = allocf(131072LL * 128);
  float*    big     = allocf((long long)B_ * NPT * NPT);   // fp32 scores, later multi(16384x1408)
  _Float16* hA      = alloch(24LL * 1024 * 1024);          // packed activations (max 16384x1408)
  _Float16* hB      = alloch(1200LL * 1024);               // packed transposed weights (max 1024x512)
  _Float16* scoresh = alloch((long long)B_ * NPT * NPT);   // fp16 softmaxed scores
  _Float16* qh      = alloch((long long)B_ * NPT * 32);
  _Float16* khb     = alloch((long long)B_ * NPT * 32);
  _Float16* vth     = alloch((long long)B_ * 128 * NPT);
  int*      fps1    = alloci(B_ * 512);
  int*      gidx1   = alloci(B_ * 512 * 32);
  float*    l1xyz   = allocf(B_ * 512 * 3);
  float*    l1pts   = allocf(B_ * 512 * 128);
  int*      fps2    = alloci(B_ * 128);
  int*      gidx2   = alloci(B_ * 128 * 64);
  float*    l2xyz   = allocf(B_ * 128 * 3);
  float*    l2sa    = allocf(B_ * 128 * 256);
  float*    l2fp    = allocf(B_ * 128 * 256);
  float*    l1fp    = allocf(B_ * 512 * 128);
  float*    l3      = allocf(B_ * 1024);
  float*    sescale = allocf(B_ * 1024);
  float*    l0feat  = allocf((long long)B_ * NPT * 128);
  float*    fbuf    = allocf((long long)B_ * NPT * 128);
  float*    qf      = allocf((long long)B_ * NPT * 16);
  float*    kf      = allocf((long long)B_ * NPT * 16);
  float*    vf      = allocf((long long)B_ * NPT * 128);
  float*    attnout = allocf((long long)B_ * NPT * 128);
  float*    featpa  = allocf((long long)B_ * NPT * 128);
  float*    fusedb  = allocf((long long)B_ * NPT * 128);
  float*    sharedb = allocf((long long)B_ * NPT * 256);
  float*    headh   = allocf((long long)B_ * NPT * 128);

  // raw GEMM on pre-packed fp16 operands
  auto gemm_raw = [&](int epi, const _Float16* A, const _Float16* Bt,
                      const float* bias, const float* g, const float* be, float* C,
                      long long M, int N, int Kp, int lda, int ldb, int ldc,
                      int batch, long long sA, long long sB, long long sC) {
    dim3 gr((N + 31) / 32, (int)(M / 16), batch), bl(32, 1, 1);
    switch (epi) {
      case 0: wmma_gemm_kernel<0><<<gr,bl,0,stream>>>(A,Bt,bias,g,be,C,N,Kp,lda,ldb,ldc,sA,sB,sC); break;
      case 1: wmma_gemm_kernel<1><<<gr,bl,0,stream>>>(A,Bt,bias,g,be,C,N,Kp,lda,ldb,ldc,sA,sB,sC); break;
      case 2: wmma_gemm_kernel<2><<<gr,bl,0,stream>>>(A,Bt,bias,g,be,C,N,Kp,lda,ldb,ldc,sA,sB,sC); break;
      case 3: wmma_gemm_kernel<3><<<gr,bl,0,stream>>>(A,Bt,bias,g,be,C,N,Kp,lda,ldb,ldc,sA,sB,sC); break;
    }
  };
  // pack fp32 A + fp32 weight, then GEMM
  auto gemm_pack = [&](int epi, const float* Af, int srcldA, int K, const float* W,
                       const float* bias, const float* g, const float* be,
                       float* C, long long M, int N) {
    const int Kp = pad32(K), Np = pad32(N);
    pack_a_kernel<<<blks(M * Kp), 256, 0, stream>>>(Af, srcldA, K, hA, Kp, M * Kp);
    pack_wt_kernel<<<blks((long long)Np * Kp), 256, 0, stream>>>(W, K, N, hB, Kp, (long long)Np * Kp);
    gemm_raw(epi, hA, hB, bias, g, be, C, M, N, Kp, Kp, Kp, N, 1, 0, 0, 0);
  };

  auto se = [&](float* data, int n, int C, int h, const float* w1, const float* b1,
                const float* w2, const float* b2) {
    se_scale_kernel<<<B_, 256, 0, stream>>>(data, n, C, h, w1, b1, w2, b2, sescale);
    long long tot = (long long)B_ * n * C;
    se_apply_kernel<<<blks(tot), 256, 0, stream>>>(data, sescale, n, C, tot);
  };
  auto mlp_chain = [&](const float* in0, int srcld0, long long rows, const int* dims, int nl,
                       const float* const* W, const float* const* bb,
                       const float* const* gg, const float* const* ee, float* finalOut) {
    const float* cur = in0; int curld = srcld0;
    for (int l = 0; l < nl; ++l) {
      float* outp = (l == nl - 1) ? finalOut : ((l & 1) ? ping : pong);
      gemm_pack(2, cur, curld, dims[l], W[l], bb[l], gg[l], ee[l], outp, rows, dims[l+1]);
      cur = outp; curld = dims[l+1];
    }
  };

  // ================= SA1 =================
  fps_kernel<<<B_, 256, 0, stream>>>(x, 6, NPT, 512, fps1);
  gather_xyz_kernel<<<blks(B_*512), 256, 0, stream>>>(x, 6, NPT, 512, fps1, l1xyz);
  ballq_kernel<<<blks(B_*512), 256, 0, stream>>>(x, 6, l1xyz, NPT, 512, 32, 0.04f, gidx1);
  group_kernel<<<blks((long long)B_*512*32), 256, 0, stream>>>(x, 6, x, 6, 6, l1xyz, gidx1, NPT, 512, 32, ping);
  { const int d[4] = {9, 64, 64, 128};
    mlp_chain(ping, 9, 131072, d, 3, s1W, s1b, s1g, s1e, pong); }
  maxpool_kernel<<<blks((long long)B_*512*128), 256, 0, stream>>>(pong, (long long)B_*512, 32, 128, l1pts);
  se(l1pts, 512, 128, 8, s1w1, s1b1, s1w2, s1b2);

  // ================= SA2 =================
  fps_kernel<<<B_, 256, 0, stream>>>(l1xyz, 3, 512, 128, fps2);
  gather_xyz_kernel<<<blks(B_*128), 256, 0, stream>>>(l1xyz, 3, 512, 128, fps2, l2xyz);
  ballq_kernel<<<blks(B_*128), 256, 0, stream>>>(l1xyz, 3, l2xyz, 512, 128, 64, 0.16f, gidx2);
  group_kernel<<<blks((long long)B_*128*64), 256, 0, stream>>>(l1xyz, 3, l1pts, 128, 128, l2xyz, gidx2, 512, 128, 64, ping);
  { const int d[4] = {131, 128, 128, 256};
    mlp_chain(ping, 131, 65536, d, 3, s2W, s2b, s2g, s2e, pong); }
  maxpool_kernel<<<blks((long long)B_*128*256), 256, 0, stream>>>(pong, (long long)B_*128, 64, 256, l2sa);
  se(l2sa, 128, 256, 16, s2w1, s2b1, s2w2, s2b2);

  // ================= SA3 (global) =================
  copy_mat_kernel<<<blks((long long)B_*128*3), 256, 0, stream>>>(l2xyz, 3, 0, ping, 259, 0, (long long)B_*128, 3);
  copy_mat_kernel<<<blks((long long)B_*128*256), 256, 0, stream>>>(l2sa, 256, 0, ping, 259, 3, (long long)B_*128, 256);
  { const int d[4] = {259, 256, 512, 1024};
    mlp_chain(ping, 259, 1024, d, 3, s3W, s3b, s3g, s3e, pong); }
  maxpool_kernel<<<blks((long long)B_*1024), 256, 0, stream>>>(pong, (long long)B_, 128, 1024, l3);
  se(l3, 1, 1024, 64, s3w1, s3b1, s3w2, s3b2);
  se(l3, 1, 1024, 64, sgw1, sgb1, sgw2, sgb2);   // se_global

  // ================= FP3 =================
  copy_mat_kernel<<<blks((long long)B_*128*256), 256, 0, stream>>>(l2sa, 256, 0, ping, 1280, 0, (long long)B_*128, 256);
  broadcast_kernel<<<blks((long long)B_*128*1024), 256, 0, stream>>>(l3, ping, 1280, 256, 128, 1024);
  { const int d[3] = {1280, 256, 256};
    mlp_chain(ping, 1280, 1024, d, 2, f3W, f3b, f3g, f3e, l2fp); }
  se(l2fp, 128, 256, 16, f3w1, f3b1, f3w2, f3b2);
  se(l2fp, 128, 256, 32, smw1, smb1, smw2, smb2); // se_mid

  // ================= FP2 =================
  copy_mat_kernel<<<blks((long long)B_*512*128), 256, 0, stream>>>(l1pts, 128, 0, ping, 384, 0, (long long)B_*512, 128);
  knn3_kernel<<<blks(B_*512), 256, 0, stream>>>(l1xyz, 3, 512, l2xyz, 128, l2fp, 256, ping, 384, 128);
  { const int d[4] = {384, 256, 256, 128};
    mlp_chain(ping, 384, 4096, d, 3, f2W, f2b, f2g, f2e, l1fp); }
  se(l1fp, 512, 128, 8, f2w1, f2b1, f2w2, f2b2);

  // ================= FP1 =================
  copy_mat_kernel<<<blks((long long)B_*NPT*6), 256, 0, stream>>>(x, 6, 0, ping, 134, 0, (long long)B_*NPT, 6);
  knn3_kernel<<<blks(B_*NPT), 256, 0, stream>>>(x, 6, NPT, l1xyz, 512, l1fp, 128, ping, 134, 6);
  { const int d[4] = {134, 128, 128, 128};
    mlp_chain(ping, 134, 16384, d, 3, f1W, f1b, f1g, f1e, l0feat); }
  se(l0feat, NPT, 128, 8, f1w1, f1b1, f1w2, f1b2);

  // ================= position attention =================
  const long long rows0 = (long long)B_ * NPT;     // 16384
  posenc_kernel<<<blks(rows0*128), 256, 0, stream>>>(l0feat, x, Wp, bp, fbuf, rows0*128);
  gemm_pack(0, fbuf, 128, 128, Wq, nullptr, nullptr, nullptr, qf, rows0, 16);
  gemm_pack(0, fbuf, 128, 128, Wk, nullptr, nullptr, nullptr, kf, rows0, 16);
  gemm_pack(0, fbuf, 128, 128, Wv, nullptr, nullptr, nullptr, vf, rows0, 128);
  pack_a_kernel<<<blks(rows0*32), 256, 0, stream>>>(qf, 16, 16, qh, 32, rows0*32);
  pack_a_kernel<<<blks(rows0*32), 256, 0, stream>>>(kf, 16, 16, khb, 32, rows0*32);
  pack_vt_kernel<<<blks((long long)B_*128*NPT), 256, 0, stream>>>(vf, vth, (long long)B_*128*NPT);
  // scores[b] = q[b] @ k[b]^T  (k already N-major: [n][kdim])
  gemm_raw(0, qh, khb, nullptr, nullptr, nullptr, big,
           NPT, NPT, 32, 32, 32, NPT, B_, (long long)NPT*32, (long long)NPT*32, (long long)NPT*NPT);
  softmax_rows_kernel<<<B_*NPT, 256, 0, stream>>>(big, NPT);
  pack_a_kernel<<<blks(rows0*NPT), 256, 0, stream>>>(big, NPT, NPT, scoresh, NPT, rows0*NPT);
  gemm_raw(0, scoresh, vth, nullptr, nullptr, nullptr, attnout,
           NPT, 128, NPT, NPT, NPT, 128, B_, (long long)NPT*NPT, 128LL*NPT, (long long)NPT*128);
  gamma_res_kernel<<<blks(rows0*128), 256, 0, stream>>>(attnout, l0feat, gamma, featpa, rows0*128);

  // ================= multi-scale fuse =================
  float* multi = big;   // scores no longer needed; 16384 x 1408 fp32
  copy_mat_kernel<<<blks(rows0*128), 256, 0, stream>>>(featpa, 128, 0, multi, 1408, 0, rows0, 128);
  broadcast_kernel<<<blks(rows0*1024), 256, 0, stream>>>(l3, multi, 1408, 128, NPT, 1024);
  lininterp_kernel<<<blks(rows0*256), 256, 0, stream>>>(l2fp, 128, 256, multi, 1408, 1152, NPT);
  gemm_pack(2, multi, 1408, 1408, fuW, fub, fug, fue, fusedb, rows0, 128);
  gemm_pack(2, fusedb, 128, 128, shW, shb, shg, she, sharedb, rows0, 256);

  // ================= heads (seg, stack, inst, lh) =================
  float* outp = (float*)d_out;
  float* hout[4]; int hco[4] = {2, 16, 16, 1}; int hepi[4] = {1, 1, 1, 3};
  hout[0] = outp;
  hout[1] = hout[0] + rows0 * 2;
  hout[2] = hout[1] + rows0 * 16;
  hout[3] = hout[2] + rows0 * 16;
  for (int hd = 0; hd < 4; ++hd) {
    gemm_pack(2, sharedb, 256, 256, hW1[hd], hb1[hd], hg1[hd], he1[hd], headh, rows0, 128);
    gemm_pack(hepi[hd], headh, 128, 128, hW2[hd], hb2[hd], nullptr, nullptr, hout[hd], rows0, hco[hd]);
  }
}
